// AttentionLayer_59193239273684
// MI455X (gfx1250) — matrix operand order repaired and checked by reading, not compiled, not run on hardware
//
#include <hip/hip_runtime.h>
#include <hip/hip_bf16.h>
#include <math.h>

#define DIM    1024
#define HEADS  16
#define DK     64
#define NTOK   64
#define GRP_PER_WG 2
#define ROWS   (NTOK * GRP_PER_WG)     // 128 token rows per WG
#define EPS_LN 1e-6f

// LDS strides (elements)
#define X2_STRIDE 1032   // 1024 + 8 halves pad -> 4-bank rotation per row
#define QK_STRIDE 72     // 64 + 8 halves pad
#define S_STRIDE  68     // 64 + 4 floats pad

// LDS layout (bytes)
#define X2_BYTES   (ROWS * X2_STRIDE * 2)          // 264192
#define QB_OFF     (X2_BYTES)                      // 264192
#define QB_BYTES   (ROWS * QK_STRIDE * 2)          // 18432 (2 groups)
#define KB_OFF     (QB_OFF + QB_BYTES)             // 282624
#define S_OFF      (KB_OFF + QB_BYTES)             // 301056
#define S_BYTES    (64 * S_STRIDE * 4)             // 17408 (one group at a time)
#define MASK_OFF   (S_OFF + S_BYTES)               // 318464
#define OACC_OFF   (MASK_OFF + ROWS * 4)           // 318976
#define SMEM_BYTES (OACC_OFF + ROWS * 4)           // 319488  (< 320KB/WGP)

typedef __attribute__((ext_vector_type(8)))  __bf16 v8bf;
typedef __attribute__((ext_vector_type(16))) __bf16 v16bf;
typedef __attribute__((ext_vector_type(8)))  float  v8f;

union Frag16 { v16bf v; v8bf h[2]; };

__device__ __forceinline__ unsigned short f32_to_bf16(float f) {
    union { float f; unsigned u; } cv; cv.f = f;
    unsigned b = cv.u;
    b += 0x7FFFu + ((b >> 16) & 1u);   // round-to-nearest-even
    return (unsigned short)(b >> 16);
}

// ---------------- weight fp32 -> bf16 conversion (runs once per launch) ----
__global__ void convert_weights_kernel(const float* __restrict__ Wq,
                                       const float* __restrict__ Wk,
                                       unsigned short* __restrict__ out) {
    const int total = DIM * DIM;
    int e = (blockIdx.x * blockDim.x + threadIdx.x) * 4;
    if (e >= 2 * total) return;
    const float* src = (e < total) ? (Wq + e) : (Wk + (e - total));
    float4 v = *(const float4*)src;
    union { ushort4 u4; unsigned short s[4]; } o;
    o.s[0] = f32_to_bf16(v.x);
    o.s[1] = f32_to_bf16(v.y);
    o.s[2] = f32_to_bf16(v.z);
    o.s[3] = f32_to_bf16(v.w);
    *(ushort4*)(out + e) = o.u4;
}

// ------- fused layernorm + QK projection + attention diag, 2 groups/WG -----
__global__ __launch_bounds__(256)
void attn_fused_kernel(const float* __restrict__ x,
                       const int* __restrict__ mask,
                       const float* __restrict__ alpha,
                       const float* __restrict__ bias,
                       const float* __restrict__ bq,
                       const float* __restrict__ bk,
                       const unsigned short* __restrict__ Wq_bf,
                       const unsigned short* __restrict__ Wk_bf,
                       float* __restrict__ out) {
    extern __shared__ unsigned char smem[];
    unsigned short* X2     = (unsigned short*)smem;
    unsigned short* Qb     = (unsigned short*)(smem + QB_OFF);
    unsigned short* Kb     = (unsigned short*)(smem + KB_OFF);
    float*          S      = (float*)(smem + S_OFF);
    float*          maskF  = (float*)(smem + MASK_OFF);
    float*          outAcc = (float*)(smem + OACC_OFF);

    const int tid  = threadIdx.x;
    const int wave = tid >> 5;
    const int lane = tid & 31;
    const int half = lane >> 4;   // 0: lanes 0-15, 1: lanes 16-31
    const int lm   = lane & 15;
    const int blk  = blockIdx.x;                      // handles groups 2b, 2b+1
    const float* xg = x + (size_t)blk * (ROWS * DIM);

    if (tid < ROWS) {
        maskF[tid]  = (mask[blk * ROWS + tid] != 0) ? 1.0f : 0.0f;
        outAcc[tid] = 0.0f;
    }

    // ---- Phase 1: layernorm (ddof=1) -> bf16 X2 in LDS (128 rows) --------
    for (int r = wave * 16; r < wave * 16 + 16; ++r) {
        const float* xr = xg + r * DIM;
        float4 xv[8];
        float s = 0.0f, s2 = 0.0f;
        #pragma unroll
        for (int j = 0; j < 8; ++j) {
            xv[j] = *(const float4*)(xr + j * 128 + lane * 4);
            s  += xv[j].x + xv[j].y + xv[j].z + xv[j].w;
            s2 += xv[j].x * xv[j].x + xv[j].y * xv[j].y +
                  xv[j].z * xv[j].z + xv[j].w * xv[j].w;
        }
        #pragma unroll
        for (int m = 16; m >= 1; m >>= 1) {
            s  += __shfl_xor(s,  m, 32);
            s2 += __shfl_xor(s2, m, 32);
        }
        float mean = s * (1.0f / 1024.0f);
        float var  = fmaxf((s2 - 1024.0f * mean * mean) * (1.0f / 1023.0f), 0.0f);
        float inv  = 1.0f / (sqrtf(var) + EPS_LN);
        unsigned short* dst = X2 + r * X2_STRIDE;
        #pragma unroll
        for (int j = 0; j < 8; ++j) {
            int d = j * 128 + lane * 4;
            float4 av = *(const float4*)(alpha + d);
            float4 bv = *(const float4*)(bias  + d);
            union { ushort4 u4; unsigned short h[4]; } o;
            o.h[0] = f32_to_bf16(av.x * (xv[j].x - mean) * inv + bv.x);
            o.h[1] = f32_to_bf16(av.y * (xv[j].y - mean) * inv + bv.y);
            o.h[2] = f32_to_bf16(av.z * (xv[j].z - mean) * inv + bv.z);
            o.h[3] = f32_to_bf16(av.w * (xv[j].w - mean) * inv + bv.w);
            *(ushort4*)(dst + d) = o.u4;
        }
    }
    __syncthreads();

    // Roles: waves 0-3 build Q tiles (mt=wave), waves 4-7 build K tiles.
    const int isK = wave >> 2;
    const int mt2 = wave & 3;
    const unsigned short* Wmat  = isK ? Wk_bf : Wq_bf;
    const float*          bvec  = isK ? bk : bq;
    unsigned short*       QKdst = isK ? Kb : Qb;

    for (int hd = 0; hd < HEADS; ++hd) {
        // ---- Phase 2: Q_h / K_h for BOTH groups, software-pipelined ------
        {
            const unsigned short* Arow[GRP_PER_WG];
            #pragma unroll
            for (int g = 0; g < GRP_PER_WG; ++g)
                Arow[g] = X2 + (g * NTOK + mt2 * 16 + lm) * X2_STRIDE + half * 8;
            const unsigned short* Brow[4];
            #pragma unroll
            for (int t = 0; t < 4; ++t)
                Brow[t] = Wmat + (size_t)(hd * 64 + t * 16 + lm) * DIM + half * 8;

            v8f acc[GRP_PER_WG][4] = {};
            Frag16 aC[GRP_PER_WG], bC[4];
            #pragma unroll
            for (int g = 0; g < GRP_PER_WG; ++g) {
                aC[g].h[0] = *(const v8bf*)(Arow[g]);        // K 0-7  / 8-15
                aC[g].h[1] = *(const v8bf*)(Arow[g] + 16);   // K 16-23 / 24-31
            }
            #pragma unroll
            for (int t = 0; t < 4; ++t) {
                bC[t].h[0] = *(const v8bf*)(Brow[t]);        // K 0-7  / 8-15
                bC[t].h[1] = *(const v8bf*)(Brow[t] + 16);   // K 16-23 / 24-31
            }

            #pragma unroll 2
            for (int k = 0; k < DIM; k += 32) {
                // Prefetch next k-step fragments (wraps to 0 on last iter;
                // harmless loads, values unused).
                const int kn = (k + 32) & (DIM - 1);
                Frag16 aN[GRP_PER_WG], bN[4];
                #pragma unroll
                for (int g = 0; g < GRP_PER_WG; ++g) {
                    aN[g].h[0] = *(const v8bf*)(Arow[g] + kn);
                    aN[g].h[1] = *(const v8bf*)(Arow[g] + kn + 16);
                }
                #pragma unroll
                for (int t = 0; t < 4; ++t) {
                    bN[t].h[0] = *(const v8bf*)(Brow[t] + kn);
                    bN[t].h[1] = *(const v8bf*)(Brow[t] + kn + 16);
                }
                // Compute on current fragments while next loads are in flight
                #pragma unroll
                for (int t = 0; t < 4; ++t)
                    #pragma unroll
                    for (int g = 0; g < GRP_PER_WG; ++g)
                        acc[g][t] = __builtin_amdgcn_wmma_f32_16x16x32_bf16(
                            false, aC[g].v, false, bC[t].v, (short)0, acc[g][t],
                            false, false);
                #pragma unroll
                for (int g = 0; g < GRP_PER_WG; ++g) aC[g] = aN[g];
                #pragma unroll
                for (int t = 0; t < 4; ++t) bC[t] = bN[t];
            }

            #pragma unroll
            for (int t = 0; t < 4; ++t) {
                int col = t * 16 + lm;
                float bb = bvec[hd * 64 + col];
                #pragma unroll
                for (int g = 0; g < GRP_PER_WG; ++g) {
                    #pragma unroll
                    for (int v = 0; v < 8; ++v) {
                        int row = g * NTOK + mt2 * 16 + v + half * 8; // C/D: VGPR v -> M=v / v+8
                        QKdst[row * QK_STRIDE + col] = f32_to_bf16(acc[g][t][v] + bb);
                    }
                }
            }
            // Prefetch next head's weight rows (global_prefetch_b8 path)
            if (hd + 1 < HEADS) {
                #pragma unroll
                for (int t = 0; t < 4; ++t)
                    __builtin_prefetch(Wmat + (size_t)((hd + 1) * 64 + t * 16 + lm) * DIM
                                            + half * 16, 0, 3);
            }
        }
        __syncthreads();

        for (int g = 0; g < GRP_PER_WG; ++g) {
            // ---- Phase 3: S = Q_h K_h^T / 8 with mask (group g) ----------
            {
                int m0    = (wave >> 1) * 16;
                int nbase = (wave & 1) * 2;
                const unsigned short* Qrow =
                    Qb + (g * NTOK + m0 + lm) * QK_STRIDE + half * 8;
                const unsigned short* Krow[2];
                #pragma unroll
                for (int t = 0; t < 2; ++t)
                    Krow[t] = Kb + (g * NTOK + (nbase + t) * 16 + lm) * QK_STRIDE + half * 8;

                v8f sacc[2] = {};
                #pragma unroll
                for (int k = 0; k < DK; k += 32) {
                    Frag16 a;
                    a.h[0] = *(const v8bf*)(Qrow + k);
                    a.h[1] = *(const v8bf*)(Qrow + k + 16);
                    #pragma unroll
                    for (int t = 0; t < 2; ++t) {
                        Frag16 b;
                        b.h[0] = *(const v8bf*)(Krow[t] + k);
                        b.h[1] = *(const v8bf*)(Krow[t] + k + 16);
                        sacc[t] = __builtin_amdgcn_wmma_f32_16x16x32_bf16(
                            false, a.v, false, b.v, (short)0, sacc[t], false, false);
                    }
                }
                #pragma unroll
                for (int t = 0; t < 2; ++t) {
                    int j = (nbase + t) * 16 + lm;
                    float mj = maskF[g * NTOK + j];
                    #pragma unroll
                    for (int v = 0; v < 8; ++v) {
                        int i = m0 + v + half * 8;
                        float val = sacc[t][v] * 0.125f;   // 1/sqrt(64)
                        float ok = maskF[g * NTOK + i] * mj;
                        S[i * S_STRIDE + j] = (ok != 0.0f) ? val : -10000.0f;
                    }
                }
            }
            __syncthreads();

            // ---- Phase 4: row softmax, accumulate diagonal (group g) -----
            {
                int r  = tid >> 2;          // 4 lanes per row, 64 rows
                int qd = tid & 3;
                const float* Srow = S + r * S_STRIDE + qd * 16;
                float mx = -1e30f;
                #pragma unroll
                for (int j = 0; j < 16; ++j) mx = fmaxf(mx, Srow[j]);
                mx = fmaxf(mx, __shfl_xor(mx, 1, 32));
                mx = fmaxf(mx, __shfl_xor(mx, 2, 32));
                float se = 0.0f;
                #pragma unroll
                for (int j = 0; j < 16; ++j) se += expf(Srow[j] - mx);
                se += __shfl_xor(se, 1, 32);
                se += __shfl_xor(se, 2, 32);
                if (qd == 0)
                    outAcc[g * NTOK + r] += expf(S[r * S_STRIDE + r] - mx) / se;
            }
            __syncthreads();
        }
    }

    if (tid < ROWS)
        out[(size_t)blk * ROWS + tid] = outAcc[tid];
}

extern "C" void kernel_launch(void* const* d_in, const int* in_sizes, int n_in,
                              void* d_out, int out_size, void* d_ws, size_t ws_size,
                              hipStream_t stream) {
    const float* x     = (const float*)d_in[0];
    const int*   mask  = (const int*)d_in[1];
    const float* alpha = (const float*)d_in[2];
    const float* bias  = (const float*)d_in[3];
    const float* Wq    = (const float*)d_in[4];
    const float* bq    = (const float*)d_in[5];
    const float* Wk    = (const float*)d_in[6];
    const float* bk    = (const float*)d_in[7];
    float* out = (float*)d_out;

    unsigned short* Wq_bf = (unsigned short*)d_ws;
    unsigned short* Wk_bf = Wq_bf + DIM * DIM;

    // Convert both weight matrices to bf16 in workspace (4 MB, L2-resident).
    {
        int total4 = (2 * DIM * DIM) / 4;
        convert_weights_kernel<<<(total4 + 255) / 256, 256, 0, stream>>>(Wq, Wk, Wq_bf);
    }

    int groups = in_sizes[0] / (NTOK * DIM);          // bs*sq = 512
    int blocks = groups / GRP_PER_WG;                 // 256 WGs
    (void)hipFuncSetAttribute((const void*)attn_fused_kernel,
                              hipFuncAttributeMaxDynamicSharedMemorySize,
                              (int)SMEM_BYTES);
    attn_fused_kernel<<<blocks, 256, SMEM_BYTES, stream>>>(
        x, mask, alpha, bias, bq, bk, Wq_bf, Wk_bf, out);
}
